// L2OOptimizer_37589553775001
// MI455X (gfx1250) — matrix operand order repaired and checked by reading, not compiled, not run on hardware
//
#include <hip/hip_runtime.h>
#include <hip/hip_bf16.h>

typedef _Float16 v16h __attribute__((ext_vector_type(16)));
typedef _Float16 v8h  __attribute__((ext_vector_type(8)));
typedef float    v8f  __attribute__((ext_vector_type(8)));

__device__ __forceinline__ float fsig(float x) {
    // sigmoid(x) = 1 / (1 + 2^(-x*log2 e))
    float e = __builtin_amdgcn_exp2f(-1.442695041f * x);
    return __builtin_amdgcn_rcpf(1.0f + e);
}
__device__ __forceinline__ float fth(float x) {
    // tanh(x) = 2*sigmoid(2x) - 1
    float e = __builtin_amdgcn_exp2f(-2.885390082f * x);
    return 2.0f * __builtin_amdgcn_rcpf(1.0f + e) - 1.0f;
}

__device__ __forceinline__ v8f wmma16(v16h a, v16h b, v8f c) {
    return __builtin_amdgcn_wmma_f32_16x16x32_f16(false, a, false, b, (short)0, c, false, false);
}

// Per-wave 16-row tile. A-frag layout (16-bit 16x32): lane L holds row M=L&15,
// halves 0..7 = K = kbA..kbA+7, halves 8..15 = K = 16+kbA..16+kbA+7, kbA = (L<16)?0:8.
// B-frag layout (16-bit 32x16): lane L holds col N=L&15, halves h -> same K as A-frag half h.
// C/D f32 16x16: lane L, vgpr r -> (M = r + ((L<16)?0:8), N = L&15).

#define HSTRIDE 56   // halves per row in transpose scratch (112B, multiple of 16B)

__global__ __launch_bounds__(256)
void l2o_lstm_kernel(const float* __restrict__ x,   const float* __restrict__ grad,
                     const float* __restrict__ h0,  const float* __restrict__ c0,
                     const float* __restrict__ h1,  const float* __restrict__ c1,
                     const float* __restrict__ W1,  const float* __restrict__ b1,
                     const float* __restrict__ W2,  const float* __restrict__ b2,
                     const float* __restrict__ Wih0, const float* __restrict__ Whh0,
                     const float* __restrict__ bih0, const float* __restrict__ bhh0,
                     const float* __restrict__ Wih1, const float* __restrict__ Whh1,
                     const float* __restrict__ bih1, const float* __restrict__ bhh1,
                     const float* __restrict__ Wfc,  const float* __restrict__ bfc,
                     float* __restrict__ out, int BN)
{
    // ---- LDS staging -------------------------------------------------------
    __shared__ __align__(32) _Float16 s_wfrag[32 * 32 * 16];   // 32 B-frags, 32KB
    __shared__ __align__(16) _Float16 s_hscr[8 * 16 * HSTRIDE]; // per-wave transpose scratch
    __shared__ float s_bias0[128], s_bias1[128];
    __shared__ float s_W1[20], s_b1[10], s_W2[100], s_b2[10], s_wfc[32];
    __shared__ float s_bfc;

    const int tid  = threadIdx.x;
    const int nthr = blockDim.x;

    for (int i = tid; i < 128; i += nthr) {
        s_bias0[i] = bih0[i] + bhh0[i];
        s_bias1[i] = bih1[i] + bhh1[i];
    }
    for (int i = tid; i < 20;  i += nthr) s_W1[i] = W1[i];
    for (int i = tid; i < 10;  i += nthr) { s_b1[i] = b1[i]; s_b2[i] = b2[i]; }
    for (int i = tid; i < 100; i += nthr) s_W2[i] = W2[i];
    for (int i = tid; i < 32;  i += nthr) s_wfc[i] = Wfc[i];
    if (tid == 0) s_bfc = bfc[0];

    // Build all 32 weight B-fragments in f16: f = m*8+n, m: 0=Wih0(K10) 1=Whh0 2=Wih1 3=Whh1
    for (int idx = tid; idx < 32 * 32; idx += nthr) {
        int f = idx >> 5, l = idx & 31;
        int n = f & 7, m = f >> 3;
        const float* W; int K;
        if (m == 0)      { W = Wih0; K = 10; }
        else if (m == 1) { W = Whh0; K = 32; }
        else if (m == 2) { W = Wih1; K = 32; }
        else             { W = Whh1; K = 32; }
        int col = n * 16 + (l & 15);
        int kb  = (l < 16) ? 0 : 8;
        _Float16* dst = &s_wfrag[(f * 32 + l) * 16];
        #pragma unroll
        for (int h = 0; h < 16; ++h) {
            int k = (h < 8) ? (kb + h) : (16 + kb + (h - 8));
            dst[h] = (k < K) ? (_Float16)W[col * K + k] : (_Float16)0.0f;
        }
    }
    __syncthreads();

    // ---- per-wave tile -----------------------------------------------------
    const int wave = tid >> 5;
    const int lane = tid & 31;
    const int row0 = (blockIdx.x * 8 + wave) * 16;
    if (row0 >= BN) return;

    const int Nl   = lane & 15;
    const int moff = (lane < 16) ? 0 : 8;   // C-layout M offset
    const int kbA  = (lane < 16) ? 0 : 8;   // A-layout K base

    float* o_x  = out;
    float* o_h0 = out + (size_t)BN;
    float* o_c0 = o_h0 + (size_t)BN * 32;
    float* o_h1 = o_c0 + (size_t)BN * 32;
    float* o_c1 = o_h1 + (size_t)BN * 32;
    float* o_d  = o_c1 + (size_t)BN * 32;

    const int rowA  = row0 + Nl;
    const int rowAc = (rowA < BN) ? rowA : (BN - 1);

    // Prefetch layer-1 state tiles while layer-0 runs
    __builtin_prefetch(h1 + (size_t)row0 * 32, 0, 1);
    __builtin_prefetch(h1 + (size_t)row0 * 32 + 256, 0, 1);
    __builtin_prefetch(c1 + (size_t)row0 * 32, 0, 1);
    __builtin_prefetch(c1 + (size_t)row0 * 32 + 256, 0, 1);

    // ---- gradient preprocessing + 2->10->10 MLP (per row, lanes L and L+16 redundant)
    float g   = grad[rowAc];
    float sgn = (g > 0.0f) ? 1.0f : ((g < 0.0f) ? -1.0f : 0.0f);
    float lg  = 0.6931471806f * __builtin_amdgcn_logf(__builtin_fabsf(g) + 1e-14f);
    float z[10], tm[10];
    #pragma unroll
    for (int i = 0; i < 10; ++i)
        z[i] = fth(s_b1[i] + s_W1[2 * i] * sgn + s_W1[2 * i + 1] * lg);
    #pragma unroll
    for (int j = 0; j < 10; ++j) {
        float a = s_b2[j];
        #pragma unroll
        for (int i = 0; i < 10; ++i) a += s_W2[j * 10 + i] * z[i];
        tm[j] = a;
    }

    // A-frag of t (K=10 zero-padded to 32), constant indices per lane half
    v16h At;
    #pragma unroll
    for (int h = 0; h < 16; ++h) At[h] = (_Float16)0.0f;
    if (lane < 16) {
        #pragma unroll
        for (int h = 0; h < 8; ++h) At[h] = (_Float16)tm[h];
    } else {
        At[0] = (_Float16)tm[8];
        At[1] = (_Float16)tm[9];
    }

    // A-frag of h0: two 32B contiguous runs per lane, f32 -> f16
    v16h Ah0;
    {
        const float* bp = h0 + (size_t)rowAc * 32 + kbA;
        float4 q0 = *(const float4*)(bp);
        float4 q1 = *(const float4*)(bp + 4);
        float4 q2 = *(const float4*)(bp + 16);
        float4 q3 = *(const float4*)(bp + 20);
        Ah0[0]=(_Float16)q0.x; Ah0[1]=(_Float16)q0.y; Ah0[2]=(_Float16)q0.z; Ah0[3]=(_Float16)q0.w;
        Ah0[4]=(_Float16)q1.x; Ah0[5]=(_Float16)q1.y; Ah0[6]=(_Float16)q1.z; Ah0[7]=(_Float16)q1.w;
        Ah0[8]=(_Float16)q2.x; Ah0[9]=(_Float16)q2.y; Ah0[10]=(_Float16)q2.z; Ah0[11]=(_Float16)q2.w;
        Ah0[12]=(_Float16)q3.x; Ah0[13]=(_Float16)q3.y; Ah0[14]=(_Float16)q3.z; Ah0[15]=(_Float16)q3.w;
    }

    // ---- layer 0 gates: 8 N-tiles x (t@Wih0.T + h0@Whh0.T) + bias ----------
    v8f acc[8];
    #pragma unroll
    for (int n = 0; n < 8; ++n) {
        float bv = s_bias0[n * 16 + Nl];
        #pragma unroll
        for (int r = 0; r < 8; ++r) acc[n][r] = bv;
        v16h Bi = *(const v16h*)&s_wfrag[((0 * 8 + n) * 32 + lane) * 16];
        acc[n] = wmma16(At, Bi, acc[n]);
        v16h Bh = *(const v16h*)&s_wfrag[((1 * 8 + n) * 32 + lane) * 16];
        acc[n] = wmma16(Ah0, Bh, acc[n]);
    }

    // ---- LSTM cell 0 (i=acc0..1, f=acc2..3, g=acc4..5, o=acc6..7) ----------
    #pragma unroll
    for (int t = 0; t < 2; ++t) {
        #pragma unroll
        for (int r = 0; r < 8; ++r) {
            int row = row0 + moff + r;
            int rc  = (row < BN) ? row : (BN - 1);
            size_t off = (size_t)rc * 32 + t * 16 + Nl;
            float cv = c0[off];
            float iv = fsig(acc[0 + t][r]);
            float fv = fsig(acc[2 + t][r]);
            float gv = fth (acc[4 + t][r]);
            float ov = fsig(acc[6 + t][r]);
            float cn = fv * cv + iv * gv;
            float hn = ov * fth(cn);
            if (row < BN) { o_c0[off] = cn; o_h0[off] = hn; }
            s_hscr[(wave * 16 + moff + r) * HSTRIDE + t * 16 + Nl] = (_Float16)hn;
        }
    }
    // C-layout -> A-layout transpose through per-wave LDS scratch (in-order DS)
    __builtin_amdgcn_wave_barrier();
    asm volatile("s_wait_dscnt 0" ::: "memory");

    v16h Ah0n;
    {
        int hb = (wave * 16 + Nl) * HSTRIDE + kbA;
        v8h lo = *(const v8h*)&s_hscr[hb];
        v8h hi = *(const v8h*)&s_hscr[hb + 16];
        #pragma unroll
        for (int h = 0; h < 8; ++h) { Ah0n[h] = lo[h]; Ah0n[8 + h] = hi[h]; }
    }

    v16h Ah1;
    {
        const float* bp = h1 + (size_t)rowAc * 32 + kbA;
        float4 q0 = *(const float4*)(bp);
        float4 q1 = *(const float4*)(bp + 4);
        float4 q2 = *(const float4*)(bp + 16);
        float4 q3 = *(const float4*)(bp + 20);
        Ah1[0]=(_Float16)q0.x; Ah1[1]=(_Float16)q0.y; Ah1[2]=(_Float16)q0.z; Ah1[3]=(_Float16)q0.w;
        Ah1[4]=(_Float16)q1.x; Ah1[5]=(_Float16)q1.y; Ah1[6]=(_Float16)q1.z; Ah1[7]=(_Float16)q1.w;
        Ah1[8]=(_Float16)q2.x; Ah1[9]=(_Float16)q2.y; Ah1[10]=(_Float16)q2.z; Ah1[11]=(_Float16)q2.w;
        Ah1[12]=(_Float16)q3.x; Ah1[13]=(_Float16)q3.y; Ah1[14]=(_Float16)q3.z; Ah1[15]=(_Float16)q3.w;
    }

    // ---- layer 1 gates: h0n@Wih1.T + h1@Whh1.T + bias ----------------------
    #pragma unroll
    for (int n = 0; n < 8; ++n) {
        float bv = s_bias1[n * 16 + Nl];
        #pragma unroll
        for (int r = 0; r < 8; ++r) acc[n][r] = bv;
        v16h Bi = *(const v16h*)&s_wfrag[((2 * 8 + n) * 32 + lane) * 16];
        acc[n] = wmma16(Ah0n, Bi, acc[n]);
        v16h Bh = *(const v16h*)&s_wfrag[((3 * 8 + n) * 32 + lane) * 16];
        acc[n] = wmma16(Ah1, Bh, acc[n]);
    }

    // ---- LSTM cell 1 -------------------------------------------------------
    float hv0[8], hv1[8];
    #pragma unroll
    for (int t = 0; t < 2; ++t) {
        #pragma unroll
        for (int r = 0; r < 8; ++r) {
            int row = row0 + moff + r;
            int rc  = (row < BN) ? row : (BN - 1);
            size_t off = (size_t)rc * 32 + t * 16 + Nl;
            float cv = c1[off];
            float iv = fsig(acc[0 + t][r]);
            float fv = fsig(acc[2 + t][r]);
            float gv = fth (acc[4 + t][r]);
            float ov = fsig(acc[6 + t][r]);
            float cn = fv * cv + iv * gv;
            float hn = ov * fth(cn);
            if (row < BN) { o_c1[off] = cn; o_h1[off] = hn; }
            if (t == 0) hv0[r] = hn; else hv1[r] = hn;
        }
    }

    // ---- delta = h1n @ Wfc.T + bfc; reduce across each 16-lane half --------
    float w0 = s_wfc[Nl], w1 = s_wfc[16 + Nl];
    float sacc[8];
    #pragma unroll
    for (int r = 0; r < 8; ++r) sacc[r] = hv0[r] * w0 + hv1[r] * w1;
    #pragma unroll
    for (int m = 1; m <= 8; m <<= 1) {
        #pragma unroll
        for (int r = 0; r < 8; ++r) sacc[r] += __shfl_xor(sacc[r], m, 32);
    }
    if (Nl < 8) {
        int row = row0 + moff + Nl;
        float dv = sacc[0];
        #pragma unroll
        for (int r = 1; r < 8; ++r) dv = (Nl == r) ? sacc[r] : dv;
        dv += s_bfc;
        if (row < BN) {
            o_d[row] = dv;
            o_x[row] = x[row] + dv;
        }
    }
}

extern "C" void kernel_launch(void* const* d_in, const int* in_sizes, int n_in,
                              void* d_out, int out_size, void* d_ws, size_t ws_size,
                              hipStream_t stream) {
    const float* x    = (const float*)d_in[0];
    const float* grad = (const float*)d_in[1];
    const float* h0   = (const float*)d_in[2];
    const float* c0   = (const float*)d_in[3];
    const float* h1   = (const float*)d_in[4];
    const float* c1   = (const float*)d_in[5];
    const float* W1   = (const float*)d_in[6];
    const float* b1   = (const float*)d_in[7];
    const float* W2   = (const float*)d_in[8];
    const float* b2   = (const float*)d_in[9];
    const float* Wih0 = (const float*)d_in[10];
    const float* Whh0 = (const float*)d_in[11];
    const float* bih0 = (const float*)d_in[12];
    const float* bhh0 = (const float*)d_in[13];
    const float* Wih1 = (const float*)d_in[14];
    const float* Whh1 = (const float*)d_in[15];
    const float* bih1 = (const float*)d_in[16];
    const float* bhh1 = (const float*)d_in[17];
    const float* Wfc  = (const float*)d_in[18];
    const float* bfc  = (const float*)d_in[19];

    int BN = in_sizes[0];                 // B*N rows
    int tiles  = (BN + 15) / 16;          // 16 rows per wave
    int blocks = (tiles + 7) / 8;         // 8 waves (256 threads) per block

    l2o_lstm_kernel<<<blocks, 256, 0, stream>>>(
        x, grad, h0, c0, h1, c1, W1, b1, W2, b2,
        Wih0, Whh0, bih0, bhh0, Wih1, Whh1, bih1, bhh1,
        Wfc, bfc, (float*)d_out, BN);
}
